// MandelbrotHybridNode_35253091566233
// MI455X (gfx1250) — compile-verified, matrix operand order checked by reading, not run on hardware
//
#include <hip/hip_runtime.h>
#include <hip/hip_bf16.h>

typedef __attribute__((ext_vector_type(16))) _Float16 v16h;
typedef __attribute__((ext_vector_type(8)))  _Float16 v8h;
typedef __attribute__((ext_vector_type(2)))  _Float16 v2h;
typedef __attribute__((ext_vector_type(8)))  float    v8f;

#define DIM 512
#define MAX_ITER 100

// ---------------- async global->LDS copy (CDNA5, ASYNCcnt-tracked) ----------------
// lds = wave-relative LDS byte offset (low 32 bits of generic pointer);
// gptr = 64-bit global address. Copies 16 bytes per lane.
__device__ __forceinline__ void async_copy_b128(unsigned lds, const _Float16* gptr) {
  asm volatile("global_load_async_to_lds_b128 %0, %1, off"
               :: "v"(lds), "v"(gptr) : "memory");
}
__device__ __forceinline__ void async_copy_b128_off16(unsigned lds, const _Float16* gptr) {
  asm volatile("global_load_async_to_lds_b128 %0, %1, off offset:16"
               :: "v"(lds), "v"(gptr) : "memory");
}
__device__ __forceinline__ void wait_async0() {
  asm volatile("s_wait_asynccnt 0x0" ::: "memory");
}
__device__ __forceinline__ unsigned lds_off(const void* p) {
  return (unsigned)(size_t)p;   // generic LDS addr: low 32 bits are the LDS offset
}

// ---------------- workspace init (hist + ordered min/max) ----------------
__global__ void k_init(unsigned int* hist, unsigned int* mm) {
  int t = threadIdx.x;
  if (t < 256) hist[t] = 0u;
  if (t == 0) { mm[0] = 0xFFFFFFFFu; mm[1] = 0u; }
}

// ---------------- f32 -> f16 weight conversion WITH transpose ----------------
__global__ void k_cvt_t(const float* __restrict__ src, _Float16* __restrict__ dst,
                        int K, int N) {
  int i = blockIdx.x * blockDim.x + threadIdx.x;
  if (i < K * N) {
    int k = i / N, n = i - k * N;
    dst[(size_t)n * K + k] = (_Float16)src[i];
  }
}

// ---------------- Mandelbrot + sparse first layers (only 6 feature cols nonzero) ----------------
__global__ void k_mandel_h1(const float* __restrict__ points,
                            const float* __restrict__ We1, const float* __restrict__ be1,
                            const float* __restrict__ Wo1, const float* __restrict__ bo1,
                            _Float16* __restrict__ h1e, _Float16* __restrict__ h1o, int B) {
  __shared__ float rawS[256][8];
  const int tid = threadIdx.x;
  const int p0  = blockIdx.x * 256;
  const int p   = p0 + tid;

  float x = 0.f, y = 0.f;
  if (p < B) { x = points[2 * (size_t)p]; y = points[2 * (size_t)p + 1]; }
  float zr = 0.f, zi = 0.f; int esc = MAX_ITER; bool done = false;
  #pragma unroll 4
  for (int n = 0; n < MAX_ITER; ++n) {
    if (!done) {
      float nr = zr * zr - zi * zi + x;
      float ni = 2.f * zr * zi + y;
      zr = nr; zi = ni;
      if (nr * nr + ni * ni > 4.f) { esc = n; done = true; }
    }
  }
  rawS[tid][0] = (float)esc * (1.f / MAX_ITER);
  rawS[tid][1] = sqrtf(zr * zr + zi * zi);
  rawS[tid][2] = zr;
  rawS[tid][3] = zi;
  rawS[tid][4] = x;
  rawS[tid][5] = y;
  __syncthreads();

  const int n0 = tid * 2;
  float we[6][2], wo[6][2];
  #pragma unroll
  for (int k = 0; k < 6; ++k) {
    we[k][0] = We1[k * DIM + n0]; we[k][1] = We1[k * DIM + n0 + 1];
    wo[k][0] = Wo1[k * DIM + n0]; wo[k][1] = Wo1[k * DIM + n0 + 1];
  }
  const float beA = be1[n0], beB = be1[n0 + 1];
  const float boA = bo1[n0], boB = bo1[n0 + 1];

  const int rows = (B - p0 < 256) ? (B - p0) : 256;
  for (int r = 0; r < rows; ++r) {
    float f0 = rawS[r][0], f1 = rawS[r][1], f2 = rawS[r][2];
    float f3 = rawS[r][3], f4 = rawS[r][4], f5 = rawS[r][5];
    float eA = beA + f0*we[0][0] + f1*we[1][0] + f2*we[2][0] + f3*we[3][0] + f4*we[4][0] + f5*we[5][0];
    float eB = beB + f0*we[0][1] + f1*we[1][1] + f2*we[2][1] + f3*we[3][1] + f4*we[4][1] + f5*we[5][1];
    float oA = boA + f0*wo[0][0] + f1*wo[1][0] + f2*wo[2][0] + f3*wo[3][0] + f4*wo[4][0] + f5*wo[5][0];
    float oB = boB + f0*wo[0][1] + f1*wo[1][1] + f2*wo[2][1] + f3*wo[3][1] + f4*wo[4][1] + f5*wo[5][1];
    v2h ve; ve[0] = (_Float16)tanhf(eA); ve[1] = (_Float16)tanhf(eB);
    v2h vo; vo[0] = (_Float16)tanhf(oA); vo[1] = (_Float16)tanhf(oB);
    *(v2h*)&h1e[(size_t)(p0 + r) * DIM + n0] = ve;
    *(v2h*)&h1o[(size_t)(p0 + r) * DIM + n0] = vo;
  }
}

// ---------------- WMMA f16 GEMM with async double-buffered LDS ----------------
// C(MxN=512) = A(MxK) * W(KxN) + bias ; W pre-transposed as Wt (N x K).
// 256 threads = 8 waves; block tile 128x64; wave tile 32x32 (4 accumulators).
// Tiles stream global->LDS via global_load_async_to_lds_b128 (ASYNCcnt),
// double-buffered so tile t+1 copies while WMMAs consume tile t.
template <int K, bool TANH, bool HAS_C, bool HAS_CH, int LDCH>
__global__ void k_gemm_wmma(const _Float16* __restrict__ A,
                            const _Float16* __restrict__ Wt,
                            const float* __restrict__ bias,
                            float* __restrict__ C, _Float16* __restrict__ Ch) {
  constexpr int N = 512;
  __shared__ __align__(16) _Float16 As[2][128][40];  // 80B row stride, 16B-aligned
  __shared__ __align__(16) _Float16 Bt[2][64][40];

  const int tid  = threadIdx.x;
  const int lane = tid & 31;
  const int wave = tid >> 5;       // 0..7
  const int wr   = wave & 3;       // 32-row slab
  const int wc   = wave >> 2;      // 32-col slab
  const int m0   = blockIdx.y * 128;
  const int n0   = blockIdx.x * 64;

  // cooperative tile mapping: A tile 128x32 (32B per thread), B tile 64x32 (16B per thread)
  const int arow = tid >> 1;            // 0..127
  const int acol = (tid & 1) * 16;      // 0 or 16
  const int brow = tid >> 2;            // 0..63
  const int bcol = (tid & 3) * 8;       // 0,8,16,24

  const _Float16* ga = A  + (size_t)(m0 + arow) * K + acol;
  const _Float16* gb = Wt + (size_t)(n0 + brow) * K + bcol;

  const unsigned la0 = lds_off(&As[0][arow][acol]);
  const unsigned la1 = lds_off(&As[1][arow][acol]);
  const unsigned lb0 = lds_off(&Bt[0][brow][bcol]);
  const unsigned lb1 = lds_off(&Bt[1][brow][bcol]);

  v8f acc00 = {}, acc01 = {}, acc10 = {}, acc11 = {};

  const int half  = lane >> 4;
  const int l15   = lane & 15;
  const int mrow0 = wr * 32 + l15;
  const int mrow1 = mrow0 + 16;
  const int nrow0 = wc * 32 + l15;
  const int nrow1 = nrow0 + 16;
  const int kfo   = half * 8;           // per-lane fragment K offset

  auto compute = [&](const _Float16 (&Ab)[128][40], const _Float16 (&Bb)[64][40]) {
    v8h lo, hi;
    lo = *(const v8h*)&Ab[mrow0][kfo];
    hi = *(const v8h*)&Ab[mrow0][16 + kfo];
    v16h a0 = __builtin_shufflevector(lo, hi, 0,1,2,3,4,5,6,7,8,9,10,11,12,13,14,15);
    lo = *(const v8h*)&Ab[mrow1][kfo];
    hi = *(const v8h*)&Ab[mrow1][16 + kfo];
    v16h a1 = __builtin_shufflevector(lo, hi, 0,1,2,3,4,5,6,7,8,9,10,11,12,13,14,15);
    lo = *(const v8h*)&Bb[nrow0][kfo];
    hi = *(const v8h*)&Bb[nrow0][16 + kfo];
    v16h b0 = __builtin_shufflevector(lo, hi, 0,1,2,3,4,5,6,7,8,9,10,11,12,13,14,15);
    lo = *(const v8h*)&Bb[nrow1][kfo];
    hi = *(const v8h*)&Bb[nrow1][16 + kfo];
    v16h b1 = __builtin_shufflevector(lo, hi, 0,1,2,3,4,5,6,7,8,9,10,11,12,13,14,15);
    acc00 = __builtin_amdgcn_wmma_f32_16x16x32_f16(false, a0, false, b0, (short)0, acc00, false, false);
    acc01 = __builtin_amdgcn_wmma_f32_16x16x32_f16(false, a0, false, b1, (short)0, acc01, false, false);
    acc10 = __builtin_amdgcn_wmma_f32_16x16x32_f16(false, a1, false, b0, (short)0, acc10, false, false);
    acc11 = __builtin_amdgcn_wmma_f32_16x16x32_f16(false, a1, false, b1, (short)0, acc11, false, false);
  };

  // prologue: tile 0 -> buffer 0
  async_copy_b128(la0, ga);
  async_copy_b128_off16(la0, ga);
  async_copy_b128(lb0, gb);
  ga += 32; gb += 32;

  const int T2 = K / 64;   // tiles processed in pairs (K multiple of 64)
  for (int j = 0; j < T2; ++j) {
    wait_async0();          // my copies into buf0 landed
    __syncthreads();        // everyone's landed; prior readers of buf1 done
    async_copy_b128(la1, ga);      // tile 2j+1 -> buf1 (always exists)
    async_copy_b128_off16(la1, ga);
    async_copy_b128(lb1, gb);
    ga += 32; gb += 32;
    compute(As[0], Bt[0]);

    wait_async0();
    __syncthreads();
    if (j + 1 < T2) {       // tile 2j+2 -> buf0
      async_copy_b128(la0, ga);
      async_copy_b128_off16(la0, ga);
      async_copy_b128(lb0, gb);
      ga += 32; gb += 32;
    }
    compute(As[1], Bt[1]);
  }

  // C/D layout: vgpr v -> M = v + 8*half ; N = lane&15 (+16 cols for acc*1; +16 rows for acc1*)
  const int na = n0 + wc * 32 + l15;
  const int nb = na + 16;
  const float bna = bias[na];
  const float bnb = bias[nb];
  #pragma unroll
  for (int v = 0; v < 8; ++v) {
    const int mA = m0 + wr * 32 + v + 8 * half;
    const int mB = mA + 16;
    float v00 = acc00[v] + bna;
    float v01 = acc01[v] + bnb;
    float v10 = acc10[v] + bna;
    float v11 = acc11[v] + bnb;
    if (TANH) { v00 = tanhf(v00); v01 = tanhf(v01); v10 = tanhf(v10); v11 = tanhf(v11); }
    if (HAS_C) {
      C[(size_t)mA * N + na] = v00;
      C[(size_t)mA * N + nb] = v01;
      C[(size_t)mB * N + na] = v10;
      C[(size_t)mB * N + nb] = v11;
    }
    if (HAS_CH) {
      Ch[(size_t)mA * LDCH + na] = (_Float16)v00;
      Ch[(size_t)mA * LDCH + nb] = (_Float16)v01;
      Ch[(size_t)mB * LDCH + na] = (_Float16)v10;
      Ch[(size_t)mB * LDCH + nb] = (_Float16)v11;
    }
  }
}

// ---------------- per-row reductions: is_in_set + fractal_dimension ----------------
__global__ void k_rowstats(const float* __restrict__ escf, const float* __restrict__ bnd,
                           float* __restrict__ isin, float* __restrict__ fd, int B) {
  const int row  = blockIdx.x * (blockDim.x >> 5) + (threadIdx.x >> 5);
  const int lane = threadIdx.x & 31;
  if (row >= B) return;
  const float* er = escf + (size_t)row * DIM;
  const float* br = bnd  + (size_t)row * DIM;
  float se = 0.f, sb = 0.f;
  for (int j = lane; j < DIM; j += 32) { se += er[j]; sb += br[j]; }
  #pragma unroll
  for (int off = 16; off; off >>= 1) {
    se += __shfl_down(se, off, 32);
    sb += __shfl_down(sb, off, 32);
  }
  if (lane == 0) {
    isin[row] = (se * (1.f / DIM) > 0.5f) ? 1.f : 0.f;
    fd[row]   = 1.f / (1.f + expf(-sb * (1.f / DIM)));
  }
}

// ---------------- ordered-uint encoding for float atomic min/max ----------------
__device__ inline unsigned int f2ord(float f) {
  unsigned int u = __float_as_uint(f);
  return (u & 0x80000000u) ? ~u : (u | 0x80000000u);
}
__device__ inline float ord2f(unsigned int u) {
  u = (u & 0x80000000u) ? (u & 0x7FFFFFFFu) : ~u;
  return __uint_as_float(u);
}

__global__ void k_minmax(const float* __restrict__ d, size_t n, unsigned int* mm) {
  size_t i = (size_t)blockIdx.x * blockDim.x + threadIdx.x;
  const size_t stride = (size_t)gridDim.x * blockDim.x;
  unsigned int lmin = 0xFFFFFFFFu, lmax = 0u;
  for (; i < n; i += stride) {
    unsigned int o = f2ord(d[i]);
    lmin = (o < lmin) ? o : lmin;
    lmax = (o > lmax) ? o : lmax;
  }
  #pragma unroll
  for (int off = 16; off; off >>= 1) {
    unsigned int a = __shfl_down(lmin, off, 32);
    unsigned int b = __shfl_down(lmax, off, 32);
    lmin = (a < lmin) ? a : lmin;
    lmax = (b > lmax) ? b : lmax;
  }
  if ((threadIdx.x & 31) == 0) {
    atomicMin(&mm[0], lmin);
    atomicMax(&mm[1], lmax);
  }
}

__global__ void k_hist(const float* __restrict__ d, size_t n, const unsigned int* __restrict__ mm,
                       unsigned int* __restrict__ hist) {
  __shared__ unsigned int h[256];
  for (int t = threadIdx.x; t < 256; t += blockDim.x) h[t] = 0u;
  __syncthreads();
  const float mn = ord2f(mm[0]);
  const float mx = ord2f(mm[1]);
  const float scale = (mx > mn) ? (256.f / (mx - mn)) : 0.f;
  size_t i = (size_t)blockIdx.x * blockDim.x + threadIdx.x;
  const size_t stride = (size_t)gridDim.x * blockDim.x;
  for (; i < n; i += stride) {
    int b = (int)((d[i] - mn) * scale);
    b = (b < 0) ? 0 : ((b > 255) ? 255 : b);
    atomicAdd(&h[b], 1u);
  }
  __syncthreads();
  for (int t = threadIdx.x; t < 256; t += blockDim.x)
    if (h[t]) atomicAdd(&hist[t], h[t]);
}

__global__ void k_entropy(const unsigned int* __restrict__ hist, float* __restrict__ out) {
  __shared__ float sh[256];
  const int t = threadIdx.x;           // launched with 256 threads
  const float hv = (float)hist[t];
  sh[t] = hv;
  __syncthreads();
  for (int s = 128; s; s >>= 1) { if (t < s) sh[t] += sh[t + s]; __syncthreads(); }
  const float total = sh[0];
  __syncthreads();
  const float p = hv / total;
  sh[t] = -p * log2f(p + 1e-10f);
  __syncthreads();
  for (int s = 128; s; s >>= 1) { if (t < s) sh[t] += sh[t + s]; __syncthreads(); }
  if (t == 0) out[0] = sh[0];
}

// ---------------- launcher ----------------
extern "C" void kernel_launch(void* const* d_in, const int* in_sizes, int n_in,
                              void* d_out, int out_size, void* d_ws, size_t ws_size,
                              hipStream_t stream) {
  (void)n_in; (void)out_size; (void)ws_size;
  const float* points = (const float*)d_in[0];
  const float* We1 = (const float*)d_in[1];
  const float* be1 = (const float*)d_in[2];
  const float* We2 = (const float*)d_in[3];
  const float* be2 = (const float*)d_in[4];
  const float* Wo1 = (const float*)d_in[5];
  const float* bo1 = (const float*)d_in[6];
  const float* Wo2 = (const float*)d_in[7];
  const float* bo2 = (const float*)d_in[8];
  const float* Wb1 = (const float*)d_in[9];
  const float* bb1 = (const float*)d_in[10];
  const float* Wb2 = (const float*)d_in[11];
  const float* bb2 = (const float*)d_in[12];

  const int B = in_sizes[0] / 2;

  float* out   = (float*)d_out;
  float* o_esc = out;
  float* o_orb = out + (size_t)B * DIM;
  float* o_bnd = out + (size_t)2 * B * DIM;
  float* o_is  = out + (size_t)3 * B * DIM;
  float* o_fd  = o_is + B;
  float* o_cx  = o_fd + B;

  char* wsp = (char*)d_ws;
  auto alloc = [&](size_t bytes) -> char* {
    char* p = wsp;
    wsp += (bytes + 255) & ~(size_t)255;
    return p;
  };
  _Float16* h1e  = (_Float16*)alloc((size_t)B * DIM * sizeof(_Float16));
  _Float16* h1o  = (_Float16*)alloc((size_t)B * DIM * sizeof(_Float16));
  _Float16* comb = (_Float16*)alloc((size_t)B * 2 * DIM * sizeof(_Float16));
  _Float16* hb   = (_Float16*)alloc((size_t)B * DIM * sizeof(_Float16));
  _Float16* We2t = (_Float16*)alloc((size_t)DIM * DIM * sizeof(_Float16));
  _Float16* Wo2t = (_Float16*)alloc((size_t)DIM * DIM * sizeof(_Float16));
  _Float16* Wb1t = (_Float16*)alloc((size_t)2 * DIM * DIM * sizeof(_Float16));
  _Float16* Wb2t = (_Float16*)alloc((size_t)DIM * DIM * sizeof(_Float16));
  unsigned int* mm   = (unsigned int*)alloc(2 * sizeof(unsigned int));
  unsigned int* hist = (unsigned int*)alloc(256 * sizeof(unsigned int));

  k_init<<<1, 256, 0, stream>>>(hist, mm);

  const int nw = DIM * DIM;
  k_cvt_t<<<(nw + 255) / 256, 256, 0, stream>>>(We2, We2t, DIM, DIM);
  k_cvt_t<<<(nw + 255) / 256, 256, 0, stream>>>(Wo2, Wo2t, DIM, DIM);
  k_cvt_t<<<(2 * nw + 255) / 256, 256, 0, stream>>>(Wb1, Wb1t, 2 * DIM, DIM);
  k_cvt_t<<<(nw + 255) / 256, 256, 0, stream>>>(Wb2, Wb2t, DIM, DIM);

  k_mandel_h1<<<(B + 255) / 256, 256, 0, stream>>>(points, We1, be1, Wo1, bo1, h1e, h1o, B);

  dim3 g(DIM / 64, B / 128);
  // escape_field = h1e @ We2 + be2 ; f16 copy into combined[:, 0:512]
  k_gemm_wmma<512, false, true, true, 1024><<<g, 256, 0, stream>>>(h1e, We2t, be2, o_esc, comb);
  // orbit_field  = h1o @ Wo2 + bo2 ; f16 copy into combined[:, 512:1024]
  k_gemm_wmma<512, false, true, true, 1024><<<g, 256, 0, stream>>>(h1o, Wo2t, bo2, o_orb, comb + DIM);
  // hb = tanh(combined @ Wb1 + bb1), f16 only
  k_gemm_wmma<1024, true, false, true, 512><<<g, 256, 0, stream>>>(comb, Wb1t, bb1, nullptr, hb);
  // boundary_field = hb @ Wb2 + bb2
  k_gemm_wmma<512, false, true, false, 0><<<g, 256, 0, stream>>>(hb, Wb2t, bb2, o_bnd, nullptr);

  k_rowstats<<<(B + 7) / 8, 256, 0, stream>>>(o_esc, o_bnd, o_is, o_fd, B);

  const size_t nb = (size_t)B * DIM;
  k_minmax<<<2048, 256, 0, stream>>>(o_bnd, nb, mm);
  k_hist<<<2048, 256, 0, stream>>>(o_bnd, nb, mm, hist);
  k_entropy<<<1, 256, 0, stream>>>(hist, o_cx);
}
